// MolecularPyTorchRF_7636451852855
// MI455X (gfx1250) — matrix operand order repaired and checked by reading, not compiled, not run on hardware
//
#include <hip/hip_runtime.h>
#include <hip/hip_bf16.h>

typedef __attribute__((ext_vector_type(16))) _Float16 v16h;
typedef __attribute__((ext_vector_type(8)))  _Float16 v8h;
typedef __attribute__((ext_vector_type(8)))  float    v8f;

#define F_DIM      1024
#define T_TREES    100
#define K_PER_TREE 32
#define NT         7          // 7 tiles of 16 trees = 112 >= 100
#define NCH        32         // 1024 / 32 K-chunks
// Wprep layout: [chunk c][tile n][packet p(0..1)][lane 0..31][j 0..7]  (halves)
#define WPREP_HALVES (NCH * NT * 2 * 32 * 8)   // 114688 halves = 224 KB
#define WPREP_BYTES  (WPREP_HALVES * 2)

// ---------------- prep kernels ----------------

__global__ void zero_ws_kernel(uint32_t* wprep_u32, float* bias) {
    int i = blockIdx.x * blockDim.x + threadIdx.x;
    if (i < WPREP_HALVES / 2)            wprep_u32[i] = 0u;
    else if (i < WPREP_HALVES / 2 + 128) bias[i - WPREP_HALVES / 2] = 0.0f;
}

// Scatter the 3200 (tree, feature) weights directly into the WMMA-B fragment
// layout. Feature indices are unique within a tree (sampled without
// replacement), so every destination slot is written at most once.
__global__ void scatter_kernel(const int* __restrict__ fidx,
                               const float* __restrict__ fw,
                               _Float16* __restrict__ wprep) {
    int e = blockIdx.x * blockDim.x + threadIdx.x;
    if (e >= T_TREES * K_PER_TREE) return;
    int t  = e >> 5;            // tree
    int f  = fidx[e];           // feature in [0,1024)
    float w = fw[e];
    int c  = f >> 5;            // K-chunk
    int kk = f & 31;            // K within chunk
    // B operand (32x16, f16): lanes 0-15 hold K=0..15 for col=lane,
    // lanes 16-31 hold K=16..31 for col=lane-16; 16 halves per lane,
    // stored as two coalesced 8-half packets.
    int lane = ((kk & 16) ? 16 : 0) + (t & 15);
    int h    = kk & 15;
    int p    = h >> 3;
    int j    = h & 7;
    int n    = t >> 4;          // N-tile
    int slot = ((((c * NT + n) * 2 + p) * 32 + lane) * 8 + j);
    wprep[slot] = (_Float16)w;
}

__global__ void bias_kernel(const float* __restrict__ thr,
                            const float* __restrict__ fw,
                            float* __restrict__ bias) {
    int t = threadIdx.x;
    if (t >= T_TREES) return;
    float s = 0.0f;
    #pragma unroll
    for (int k = 0; k < K_PER_TREE; ++k)
        s += thr[t * K_PER_TREE + k] * fw[t * K_PER_TREE + k];
    bias[t] = s;
}

// ---------------- main WMMA GEMM + sigmoid-mean kernel ----------------
// Block: 256 threads = 8 waves; each wave computes 16 rows x 112 trees.
// Grid: B/128 workgroups.

__global__ __launch_bounds__(256)
void forest_wmma_kernel(const float* __restrict__ x,
                        const _Float16* __restrict__ wprep,
                        const float* __restrict__ bias,
                        float* __restrict__ out) {
    const int lane    = threadIdx.x & 31;
    const int wave    = threadIdx.x >> 5;
    const int rowBase = blockIdx.x * 128 + wave * 16;
    const int col     = lane & 15;
    const int kbase   = (lane < 16) ? 0 : 16;  // A-operand K base for this lane

    const v8f vzero = {0.f, 0.f, 0.f, 0.f, 0.f, 0.f, 0.f, 0.f};
    v8f acc[NT];
    #pragma unroll
    for (int n = 0; n < NT; ++n) acc[n] = vzero;

    // A operand (16x32, f16): lane row = lane&15,
    // K-set = kbase + {0..15} -> one contiguous
    // 16-float run in the row: four float4 loads.
    const float* xrow = x + (size_t)(rowBase + col) * F_DIM;

    #pragma unroll 2
    for (int c = 0; c < NCH; ++c) {
        const float4* ap = (const float4*)(xrow + c * 32 + kbase);
        float4 f0 = ap[0];   // K = kbase + 0..3
        float4 f1 = ap[1];   // K = kbase + 4..7
        float4 f2 = ap[2];   // K = kbase + 8..11
        float4 f3 = ap[3];   // K = kbase + 12..15
        if (c + 1 < NCH)
            __builtin_prefetch(xrow + (c + 1) * 32 + kbase, 0, 3);

        v16h a;
        a[0]  = (_Float16)f0.x; a[1]  = (_Float16)f0.y;
        a[2]  = (_Float16)f0.z; a[3]  = (_Float16)f0.w;
        a[4]  = (_Float16)f1.x; a[5]  = (_Float16)f1.y;
        a[6]  = (_Float16)f1.z; a[7]  = (_Float16)f1.w;
        a[8]  = (_Float16)f2.x; a[9]  = (_Float16)f2.y;
        a[10] = (_Float16)f2.z; a[11] = (_Float16)f2.w;
        a[12] = (_Float16)f3.x; a[13] = (_Float16)f3.y;
        a[14] = (_Float16)f3.z; a[15] = (_Float16)f3.w;

        #pragma unroll
        for (int n = 0; n < NT; ++n) {
            const v8h* bp = (const v8h*)(wprep
                              + (size_t)((c * NT + n) * 2) * 256  // 256 halves per packet-of-32-lanes
                              + lane * 8);
            v8h blo = bp[0];    // p = 0 packet (fully coalesced 16B/lane)
            v8h bhi = bp[32];   // p = 1 packet (+256 halves)
            v16h b;
            #pragma unroll
            for (int j = 0; j < 8; ++j) { b[j] = blo[j]; b[8 + j] = bhi[j]; }

            acc[n] = __builtin_amdgcn_wmma_f32_16x16x32_f16(
                /*neg_a=*/false, a, /*neg_b=*/false, b,
                /*c_mod=*/(short)0, acc[n],
                /*reuse_a=*/false, /*reuse_b=*/false);
        }
    }

    // Epilogue: D layout -> VGPR r, lane L: row M = r + (L>=16 ? 8 : 0), col N = L&15.
    float s[8];
    #pragma unroll
    for (int r = 0; r < 8; ++r) s[r] = 0.0f;

    #pragma unroll
    for (int n = 0; n < NT; ++n) {
        int tree = n * 16 + col;            // < 112; bias[100..127] zeroed
        float bn = bias[tree];
        float m  = (tree < T_TREES) ? 1.0f : 0.0f;
        #pragma unroll
        for (int r = 0; r < 8; ++r) {
            float z = acc[n][r] - bn;
            float p = 1.0f / (1.0f + __expf(-z));
            s[r] += m * p;
        }
    }

    // Reduce over the 16 lanes that share a row (xor 1,2,4,8 stays in-half).
    #pragma unroll
    for (int off = 1; off < 16; off <<= 1) {
        #pragma unroll
        for (int r = 0; r < 8; ++r)
            s[r] += __shfl_xor(s[r], off, 32);
    }

    if (col == 0) {
        float2* o2 = (float2*)out;
        int rb = rowBase + ((lane >> 4) << 3);
        #pragma unroll
        for (int r = 0; r < 8; ++r) {
            float pm = s[r] * (1.0f / (float)T_TREES);
            o2[rb + r] = make_float2(1.0f - pm, pm);
        }
    }
}

// ---------------- launcher ----------------

extern "C" void kernel_launch(void* const* d_in, const int* in_sizes, int n_in,
                              void* d_out, int out_size, void* d_ws, size_t ws_size,
                              hipStream_t stream) {
    const float* x    = (const float*)d_in[0];
    const int*   fidx = (const int*)d_in[1];
    const float* fthr = (const float*)d_in[2];
    const float* fw   = (const float*)d_in[3];
    float*       out  = (float*)d_out;

    _Float16* wprep = (_Float16*)d_ws;
    float*    bias  = (float*)((char*)d_ws + WPREP_BYTES);

    const int Brows = in_sizes[0] / F_DIM;   // 131072

    {
        int n = WPREP_HALVES / 2 + 128;
        zero_ws_kernel<<<(n + 255) / 256, 256, 0, stream>>>((uint32_t*)d_ws, bias);
    }
    scatter_kernel<<<(T_TREES * K_PER_TREE + 255) / 256, 256, 0, stream>>>(fidx, fw, wprep);
    bias_kernel<<<1, 128, 0, stream>>>(fthr, fw, bias);

    forest_wmma_kernel<<<Brows / 128, 256, 0, stream>>>(x, wprep, bias, out);
}
